// Conductor_76699525972589
// MI455X (gfx1250) — matrix operand order repaired and checked by reading, not compiled, not run on hardware
//
#include <hip/hip_runtime.h>
#include <stddef.h>

// MI455X / gfx1250: wave32, WMMA bf16 path. All GEMMs via v_wmma_f32_16x16x32_bf16.
// Staging uses GLOBAL_LOAD_ASYNC_TO_LDS_B128 (ASYNCcnt) when the builtin exists.

typedef __attribute__((ext_vector_type(16))) __bf16 v16bf;
typedef __attribute__((ext_vector_type(8)))  float  v8f;
typedef __attribute__((ext_vector_type(4)))  int    v4i;

#define HID   1024
#define BSZ   1024
#define GH    (4 * HID)   // 4096 gate width

// ---- async global->LDS support (gfx1250). Guarded: falls back to reg staging.
#if defined(__HIP_DEVICE_COMPILE__) && defined(__has_builtin)
#if __has_builtin(__builtin_amdgcn_global_load_async_to_lds_b128)
#define USE_ASYNC 1
#endif
#endif
#ifndef USE_ASYNC
#define USE_ASYNC 0
#endif

__device__ __forceinline__ void async_wait_all() {
#if USE_ASYNC
#if __has_builtin(__builtin_amdgcn_s_wait_asynccnt)
  __builtin_amdgcn_s_wait_asynccnt(0);
#else
  asm volatile("s_wait_asynccnt 0" ::: "memory");
#endif
#endif
}

// ---------------------------------------------------------------------------
// fp32 -> bf16 conversion (weights, once per launch)
// ---------------------------------------------------------------------------
__global__ __launch_bounds__(256) void cvt_f32_bf16(const float* __restrict__ src,
                                                    __bf16* __restrict__ dst, int n) {
  int i = blockIdx.x * 256 + threadIdx.x;
  if (i < n) dst[i] = (__bf16)src[i];
}

// b0 = b_ih0 + b_hh0 ; b1 = b_ih1 + b_hh1
__global__ __launch_bounds__(256) void bias_sum(const float* __restrict__ a0, const float* __restrict__ a1,
                                                const float* __restrict__ c0, const float* __restrict__ c1,
                                                float* __restrict__ o0, float* __restrict__ o1, int n) {
  int i = blockIdx.x * 256 + threadIdx.x;
  if (i < n) { o0[i] = a0[i] + a1[i]; o1[i] = c0[i] + c1[i]; }
}

// init: h0=c0=h1=c1=z ; bf16 copies of z for GEMM inputs
__global__ __launch_bounds__(256) void init_states(const float* __restrict__ z,
                                                   float* __restrict__ c0, float* __restrict__ c1,
                                                   __bf16* __restrict__ h0b, __bf16* __restrict__ h1b,
                                                   __bf16* __restrict__ zb, int n) {
  int i = blockIdx.x * 256 + threadIdx.x;
  if (i < n) {
    float v = z[i];
    c0[i] = v; c1[i] = v;
    __bf16 b = (__bf16)v;
    h0b[i] = b; h1b[i] = b; zb[i] = b;
  }
}

// ---------------------------------------------------------------------------
// GEMM: C[M,N] = A1[M,K] * W1[N,K]^T (+ A2*W2^T) (+ bias[N]) (+ addend[M,N])
// bf16 inputs, fp32 accumulate via v_wmma_f32_16x16x32_bf16.
// Block = 256 threads (8 wave32), C tile 128(M) x 128(N), K-step 64,
// double-buffered LDS (2 x 32KB). Wave = 64x32 subtile -> 4x2 fragments.
// LDS slabs pre-swizzled into WMMA fragment layout (ISA 7.12.2):
//   A frag (16x32): lane l holds row m=l&15; elem j -> k=(j&7)+16*(j>>3)+8*(l>>4)
//   B frag (32x16): lane l holds col n=l&15; elem j -> k=(j&7)+16*(j>>3)+8*(l>>4)
// so each fragment read is 32 contiguous bytes per lane (2x ds_load_b128),
// and each staged chunk is one 16B global -> 16B LDS async DMA (ASYNCcnt).
// ---------------------------------------------------------------------------
__global__ __launch_bounds__(256)
void gemm_bf16_wmma(const __bf16* __restrict__ A1, const __bf16* __restrict__ W1,
                    const __bf16* __restrict__ A2, const __bf16* __restrict__ W2,
                    const float* __restrict__ bias, const float* __restrict__ addend,
                    float* __restrict__ C, int M, int N, int K) {
  __shared__ __align__(32) unsigned char lds[2 * 32768];  // per buffer: A 16KB + B 16KB

  const int tid  = threadIdx.x;
  const int lane = tid & 31;
  const int w    = tid >> 5;      // wave id 0..7
  const int wm   = w & 1;         // 0..1 -> 64-row strip (4 m-fragments)
  const int wn   = w >> 1;        // 0..3 -> 32-col strip (2 n-fragments)
  const int mBase = blockIdx.y * 128;
  const int nBase = blockIdx.x * 128;

  v8f acc[4][2] = {};

  // ---- stage one 64-wide K slab (A: 128x64, B: 128x64) into LDS buffer ----
  auto stage = [&](const __bf16* __restrict__ A, const __bf16* __restrict__ W,
                   int kb, int buf) {
    unsigned char* base = lds + buf * 32768;
#pragma unroll
    for (int i = 0; i < 4; ++i) {           // A: 1024 chunks of 16B
      int c = tid + i * 256;
      int m = c >> 3;                       // 0..127
      int g = c & 7;                        // k-group of 8 within 64
      int s = g >> 2;                       // 32-k sub-slab
      int gg = g & 3;
      const __bf16* gp = A + (size_t)(mBase + m) * K + kb + 8 * g;
      int off = s * 8192 + ((m >> 4) << 10)
              + (((m & 15) + ((gg & 1) << 4)) << 5) + ((gg >> 1) << 4);
#if USE_ASYNC
      __builtin_amdgcn_global_load_async_to_lds_b128((v4i*)gp, (v4i*)(base + off), 0, 0);
#else
      *reinterpret_cast<uint4*>(base + off) = *reinterpret_cast<const uint4*>(gp);
#endif
    }
#pragma unroll
    for (int i = 0; i < 4; ++i) {           // B: 1024 chunks of 16B
      int c = tid + i * 256;
      int n = c >> 3;                       // 0..127
      int p = c & 7;
      int s = p >> 2;
      int pp = p & 3;
      const __bf16* gp = W + (size_t)(nBase + n) * K + kb + 8 * p;
      int off = 16384 + s * 8192 + ((n >> 4) << 10)
              + (((n & 15) + ((pp & 1) << 4)) << 5) + ((pp >> 1) << 4);
#if USE_ASYNC
      __builtin_amdgcn_global_load_async_to_lds_b128((v4i*)gp, (v4i*)(base + off), 0, 0);
#else
      *reinterpret_cast<uint4*>(base + off) = *reinterpret_cast<const uint4*>(gp);
#endif
    }
  };

  // ---- 16 WMMAs against one staged buffer ----
  auto compute = [&](int buf) {
    unsigned char* base = lds + buf * 32768;
#pragma unroll
    for (int s = 0; s < 2; ++s) {
      v16bf af[4], bfr[2];
#pragma unroll
      for (int mi = 0; mi < 4; ++mi)
        af[mi] = *reinterpret_cast<const v16bf*>(base + s * 8192 + ((wm * 4 + mi) << 10) + lane * 32);
#pragma unroll
      for (int ni = 0; ni < 2; ++ni)
        bfr[ni] = *reinterpret_cast<const v16bf*>(base + 16384 + s * 8192 + ((wn * 2 + ni) << 10) + lane * 32);
#pragma unroll
      for (int mi = 0; mi < 4; ++mi)
#pragma unroll
        for (int ni = 0; ni < 2; ++ni)
          acc[mi][ni] = __builtin_amdgcn_wmma_f32_16x16x32_bf16(
              false, af[mi], false, bfr[ni], (short)0, acc[mi][ni], false, false);
    }
  };

  const int itersPerPass = K >> 6;                 // K/64
  const int nIter = (A2 == nullptr) ? itersPerPass : 2 * itersPerPass;

  stage(A1, W1, 0, 0);
  async_wait_all();
  __syncthreads();

  for (int it = 0; it < nIter; ++it) {
    const int buf = it & 1;
    if (it + 1 < nIter) {
      const int nx = it + 1;
      const __bf16* A = (nx >= itersPerPass) ? A2 : A1;
      const __bf16* W = (nx >= itersPerPass) ? W2 : W1;
      stage(A, W, (nx >= itersPerPass ? nx - itersPerPass : nx) << 6, buf ^ 1);
    }
    compute(buf);
    async_wait_all();     // next buffer's DMA landed in LDS
    __syncthreads();      // everyone done reading current buffer
  }

  // ---- epilogue: C/D layout: lane l -> n=l&15 ; vgpr r -> m = r + 8*(l>>4) ----
#pragma unroll
  for (int mi = 0; mi < 4; ++mi) {
#pragma unroll
    for (int ni = 0; ni < 2; ++ni) {
      int n  = nBase + (wn * 2 + ni) * 16 + (lane & 15);
      int m0 = mBase + (wm * 4 + mi) * 16 + ((lane >> 4) << 3);
      float bv = bias ? bias[n] : 0.0f;
#pragma unroll
      for (int r = 0; r < 8; ++r) {
        size_t idx = (size_t)(m0 + r) * N + n;
        float v = acc[mi][ni][r] + bv;
        if (addend) v += addend[idx];
        C[idx] = v;
      }
    }
  }
}

// ---------------------------------------------------------------------------
// Pointwise LSTM cell: gates G[B,4H] (i,f,g,o), state c[B,H] (in/out),
// writes next h as bf16 (GEMM input); layer-1 also writes feat[(b*4+t), :].
// ---------------------------------------------------------------------------
__global__ __launch_bounds__(256)
void lstm_pointwise(const float* __restrict__ G, float* __restrict__ c,
                    __bf16* __restrict__ hbf, __bf16* __restrict__ featbf, int t) {
  int i = blockIdx.x * 256 + threadIdx.x;     // over B*H
  int b = i >> 10;                            // H == 1024
  int n = i & (HID - 1);
  const float* g4 = G + (size_t)b * GH + n;
  float gi = g4[0];
  float gf = g4[HID];
  float gg = g4[2 * HID];
  float go = g4[3 * HID];
  float si = 1.0f / (1.0f + __expf(-gi));
  float sf = 1.0f / (1.0f + __expf(-gf));
  float so = 1.0f / (1.0f + __expf(-go));
  float tg = tanhf(gg);
  float cn = sf * c[i] + si * tg;
  float hn = so * tanhf(cn);
  c[i] = cn;
  __bf16 hb = (__bf16)hn;
  hbf[i] = hb;
  if (featbf) featbf[(size_t)(b * 4 + t) * HID + n] = hb;
}

// ---------------------------------------------------------------------------
// Host launch sequence (all on `stream`; graph-capture safe).
// Workspace budget ~88.1 MB.
// ---------------------------------------------------------------------------
extern "C" void kernel_launch(void* const* d_in, const int* in_sizes, int n_in,
                              void* d_out, int out_size, void* d_ws, size_t ws_size,
                              hipStream_t stream) {
  (void)in_sizes; (void)n_in; (void)out_size; (void)ws_size;
  const float* z    = (const float*)d_in[0];
  const float* Wih0 = (const float*)d_in[1];
  const float* Whh0 = (const float*)d_in[2];
  const float* bih0 = (const float*)d_in[3];
  const float* bhh0 = (const float*)d_in[4];
  const float* Wih1 = (const float*)d_in[5];
  const float* Whh1 = (const float*)d_in[6];
  const float* bih1 = (const float*)d_in[7];
  const float* bhh1 = (const float*)d_in[8];
  const float* Wlin = (const float*)d_in[9];
  const float* blin = (const float*)d_in[10];
  float* out = (float*)d_out;

  char* ws = (char*)d_ws;
  size_t off = 0;
  auto carve = [&](size_t bytes) { char* p = ws + off; off += (bytes + 255) & ~(size_t)255; return p; };

  const size_t NW  = (size_t)GH * HID;   // 4M elems per gate weight
  const size_t NBH = (size_t)BSZ * HID;  // 1M state elems

  __bf16* Wih0b = (__bf16*)carve(NW * 2);
  __bf16* Whh0b = (__bf16*)carve(NW * 2);
  __bf16* Wih1b = (__bf16*)carve(NW * 2);
  __bf16* Whh1b = (__bf16*)carve(NW * 2);
  __bf16* Wlinb = (__bf16*)carve((size_t)HID * HID * 2);
  float*  b0    = (float*)carve(GH * 4);
  float*  b1    = (float*)carve(GH * 4);
  float*  Gz0   = (float*)carve((size_t)BSZ * GH * 4);   // z@Wih0^T + b0, reused all steps
  float*  Gbuf  = (float*)carve((size_t)BSZ * GH * 4);   // per-cell gate scratch
  float*  c0    = (float*)carve(NBH * 4);
  float*  c1    = (float*)carve(NBH * 4);
  __bf16* h0b   = (__bf16*)carve(NBH * 2);
  __bf16* h1b   = (__bf16*)carve(NBH * 2);
  __bf16* zb    = (__bf16*)carve(NBH * 2);
  __bf16* featb = (__bf16*)carve((size_t)BSZ * 4 * HID * 2);

  // 1) weight conversion fp32 -> bf16
  {
    int gw = (int)((NW + 255) / 256);
    cvt_f32_bf16<<<gw, 256, 0, stream>>>(Wih0, Wih0b, (int)NW);
    cvt_f32_bf16<<<gw, 256, 0, stream>>>(Whh0, Whh0b, (int)NW);
    cvt_f32_bf16<<<gw, 256, 0, stream>>>(Wih1, Wih1b, (int)NW);
    cvt_f32_bf16<<<gw, 256, 0, stream>>>(Whh1, Whh1b, (int)NW);
    cvt_f32_bf16<<<(HID * HID + 255) / 256, 256, 0, stream>>>(Wlin, Wlinb, HID * HID);
  }
  // 2) bias sums + state init
  bias_sum<<<GH / 256, 256, 0, stream>>>(bih0, bhh0, bih1, bhh1, b0, b1, GH);
  init_states<<<(int)(NBH / 256), 256, 0, stream>>>(z, c0, c1, h0b, h1b, zb, (int)NBH);

  dim3 blk(256);
  dim3 grdGates(GH / 128, BSZ / 128);   // 32 x 8 = 256 blocks

  // 3) Gz0 = z @ Wih0^T + (b_ih0 + b_hh0)   (constant across steps)
  gemm_bf16_wmma<<<grdGates, blk, 0, stream>>>(zb, Wih0b, nullptr, nullptr,
                                               b0, nullptr, Gz0, BSZ, GH, HID);

  // 4) recurrence (kernel boundaries provide grid-wide sync)
  for (int t = 0; t < 4; ++t) {
    // layer 0 gates: G = h0 @ Whh0^T + Gz0
    gemm_bf16_wmma<<<grdGates, blk, 0, stream>>>(h0b, Whh0b, nullptr, nullptr,
                                                 nullptr, Gz0, Gbuf, BSZ, GH, HID);
    lstm_pointwise<<<(int)(NBH / 256), blk, 0, stream>>>(Gbuf, c0, h0b, nullptr, 0);

    // layer 1 gates (fused dual GEMM): G = h0 @ Wih1^T + h1 @ Whh1^T + b1
    gemm_bf16_wmma<<<grdGates, blk, 0, stream>>>(h0b, Wih1b, h1b, Whh1b,
                                                 b1, nullptr, Gbuf, BSZ, GH, HID);
    lstm_pointwise<<<(int)(NBH / 256), blk, 0, stream>>>(Gbuf, c1, h1b, featb, t);
  }

  // 5) out[(b*4+t), :] = feat @ Wlin^T + b_lin   (M=4096, N=1024, K=1024)
  dim3 grdLin(HID / 128, (BSZ * 4) / 128);  // 8 x 32 = 256 blocks
  gemm_bf16_wmma<<<grdLin, blk, 0, stream>>>(featb, Wlinb, nullptr, nullptr,
                                             blin, nullptr, out, BSZ * 4, HID, HID);
}
